// HVGuardModel_44332652429432
// MI455X (gfx1250) — matrix operand order repaired and checked by reading, not compiled, not run on hardware
//
#include <hip/hip_runtime.h>

// ---------------- types / helpers ----------------
typedef __attribute__((ext_vector_type(16))) __bf16 v16bf;
typedef __attribute__((ext_vector_type(8)))  float  v8f;

union FragB16 { v16bf v; uint4 q[2]; };

__device__ __forceinline__ unsigned short f2bf(float f) {
  unsigned u = __builtin_bit_cast(unsigned, f);
  u += 0x7fffu + ((u >> 16) & 1u);          // round-to-nearest-even
  return (unsigned short)(u >> 16);
}
__device__ __forceinline__ unsigned pack2bf(float lo, float hi) {
  return (unsigned)f2bf(lo) | ((unsigned)f2bf(hi) << 16);
}
__device__ __forceinline__ float bf2f(unsigned short h) {
  unsigned u = ((unsigned)h) << 16;
  return __builtin_bit_cast(float, u);
}

#define Dd 1024
#define Ee 8
#define Hh 128
#define TM 64      // tokens per workgroup
#define XS 40      // s_x row stride (bf16 elems): 80B rows, bank-conflict free
#define HSS 264    // s_hs row stride: 528B rows
#define MXS 136    // s_mix row stride: 272B rows
#define TS 68      // s_t row stride (floats)

// ---------------- weight prep: fp32 -> bf16, [N][K] layout ----------------
__global__ void prep_weights(const float* __restrict__ We1,
                             const float* __restrict__ We2,
                             const float* __restrict__ Wc1,
                             unsigned short* __restrict__ We1t,
                             unsigned short* __restrict__ We2t,
                             unsigned short* __restrict__ Wc1t) {
  int i = blockIdx.x * blockDim.x + threadIdx.x;
  const int N1 = 1024 * 1024;     // We1t: n_all(=e*128+h) x k(D)
  const int N2 = 128 * 1024;      // We2t: n(H) x k_all(=e*128+k)
  const int N3 = 64 * 128;        // Wc1t: n(C1) x k(H)
  if (i < N1) {
    int n = i >> 10, k = i & 1023;
    int e = n >> 7, h = n & 127;
    We1t[i] = f2bf(We1[(e * 1024 + k) * 128 + h]);      // We1[e][k][h]
  } else if (i < N1 + N2) {
    int j = i - N1;
    int n = j >> 10, ka = j & 1023;                     // ka = e*128+k
    We2t[j] = f2bf(We2[ka * 128 + n]);                  // We2[e][k][n]
  } else if (i < N1 + N2 + N3) {
    int j = i - N1 - N2;
    int n = j >> 7, k = j & 127;
    Wc1t[j] = f2bf(Wc1[k * 64 + n]);                    // Wc1[k][n]
  }
}

// ---------------- fused MoE kernel ----------------
__global__ __launch_bounds__(1024) void moe_fused(
    const float* __restrict__ x,
    const float* __restrict__ be1, const float* __restrict__ be2,
    const float* __restrict__ Wg,  const float* __restrict__ bg,
    const float* __restrict__ bc1, const float* __restrict__ Wc2,
    const float* __restrict__ bc2,
    const unsigned short* __restrict__ We1t,
    const unsigned short* __restrict__ We2t,
    const unsigned short* __restrict__ Wc1t,
    float* __restrict__ out)
{
  __shared__ unsigned short s_x[2][TM * XS];  // 10  KB  x K-chunk, double buffered
  __shared__ float          s_gl[TM * 8];     //  2  KB  gate logits
  __shared__ float          s_gw[TM * 8];     //  2  KB  gate softmax
  __shared__ unsigned short s_hs[TM * HSS];   // 33  KB  gated hidden (bf16, per e-group)
  __shared__ unsigned short s_mix[TM * MXS];  // 17  KB  mixed features (bf16)
  __shared__ float          s_t[TM * TS];     // 17  KB  cls hidden

  const int tid = threadIdx.x;
  const int w   = tid >> 5;            // wave 0..31
  const int ln  = tid & 31;
  const int l15 = ln & 15;
  const int lhi = ln >> 4;             // lane-half
  const long tok0 = (long)blockIdx.x * TM;

  // x staging mapping: thread t loads one contiguous float2 of the [64 x 32] chunk
  const int xr = tid >> 4;             // row 0..63
  const int xc = (tid & 15) * 2;       // col 0,2,..,30

  // ---- stage 1: h = relu(x @ We1 + be1), M=64 N=1024 K=1024 ----
  // wave w owns N-tiles {2w, 2w+1} (cols 32w..32w+31), all 4 M-tiles.
  v8f acc1[4][2];
  #pragma unroll
  for (int mt = 0; mt < 4; ++mt)
    #pragma unroll
    for (int nt = 0; nt < 2; ++nt) acc1[mt][nt] = {};

  float greg = 0.0f;   // gate partial: thread t<512 owns (tok=t>>3, e=t&7)

  // preload chunk 0
  {
    const float2 v = *(const float2*)&x[(tok0 + xr) * Dd + xc];
    *(unsigned*)&s_x[0][xr * XS + xc] = pack2bf(v.x, v.y);
  }
  __syncthreads();

  for (int it = 0; it < Dd / 32; ++it) {
    const int kc  = it * 32;
    const int buf = it & 1;
    const bool has_next = (it + 1) < (Dd / 32);

    // issue next chunk's global loads early; they overlap the WMMAs below
    float2 nxt;
    if (has_next)
      nxt = *(const float2*)&x[(tok0 + xr) * Dd + kc + 32 + xc];

    // gate VALU work co-executes with WMMA (XDL ops tracked as TRANS)
    if (tid < 512) {
      int gt = tid >> 3, ge = tid & 7;
      #pragma unroll 8
      for (int c = 0; c < 32; ++c)
        greg += bf2f(s_x[buf][gt * XS + c]) * Wg[(kc + c) * Ee + ge];
    }

    // A fragments: lane l15 = row, lane-half selects K 0..15 / 16..31
    FragB16 a[4];
    #pragma unroll
    for (int mt = 0; mt < 4; ++mt) {
      const unsigned short* p = &s_x[buf][(mt * 16 + l15) * XS + lhi * 16];
      a[mt].q[0] = *(const uint4*)p;
      a[mt].q[1] = *(const uint4*)(p + 8);
    }
    #pragma unroll
    for (int nt = 0; nt < 2; ++nt) {
      // B fragment: lane l15 = col n, lane-half selects K 0..15 / 16..31 (contig 32B)
      int n = w * 32 + nt * 16 + l15;
      const unsigned short* p = &We1t[n * Dd + kc + lhi * 16];
      FragB16 b;
      b.q[0] = *(const uint4*)p;
      b.q[1] = *(const uint4*)(p + 8);
      if (has_next) __builtin_prefetch(p + 32, 0, 3);   // next K-chunk of this column
      #pragma unroll
      for (int mt = 0; mt < 4; ++mt)
        acc1[mt][nt] = __builtin_amdgcn_wmma_f32_16x16x32_bf16(
            false, a[mt].v, false, b.v, (short)0, acc1[mt][nt], false, false);
    }

    // stage next chunk into the other buffer (safe: prior reads of it were
    // before the previous barrier), then one barrier per iteration
    if (has_next)
      *(unsigned*)&s_x[buf ^ 1][xr * XS + xc] = pack2bf(nxt.x, nxt.y);
    __syncthreads();
  }

  // ---- gate softmax ----
  if (tid < 512) s_gl[tid] = greg + bg[tid & 7];
  __syncthreads();
  if (tid < TM) {
    float mx = -1e30f;
    #pragma unroll
    for (int e = 0; e < 8; ++e) mx = fmaxf(mx, s_gl[tid * 8 + e]);
    float ex[8], sum = 0.f;
    #pragma unroll
    for (int e = 0; e < 8; ++e) { ex[e] = __expf(s_gl[tid * 8 + e] - mx); sum += ex[e]; }
    float inv = 1.0f / sum;
    #pragma unroll
    for (int e = 0; e < 8; ++e) s_gw[tid * 8 + e] = ex[e] * inv;
  }
  __syncthreads();

  // ---- stage 2: mix = (gw .* relu(h+be1)) @ concat_e(We2[e]),  M=64 N=128 K=1024 ----
  // gating folded into A; K swept in 4 expert-group passes of 256 through s_hs.
  v8f acc2 = {};
  const int nt2 = w & 7;     // N tile 0..7
  const int mt2 = w >> 3;    // M tile 0..3
  for (int g = 0; g < 4; ++g) {
    if ((w >> 3) == g) {     // this group's waves stage their stage-1 output
      #pragma unroll
      for (int mt = 0; mt < 4; ++mt) {
        int mb = mt * 16 + lhi * 8;
        #pragma unroll
        for (int nt = 0; nt < 2; ++nt) {
          int nglob = w * 32 + nt * 16 + l15;
          int nloc  = (w & 7) * 32 + nt * 16 + l15;
          float bias = be1[nglob];
          #pragma unroll
          for (int r = 0; r < 8; ++r) {
            float v = fmaxf(acc1[mt][nt][r] + bias, 0.0f);
            v *= s_gw[(mb + r) * 8 + (nglob >> 7)];
            s_hs[(mb + r) * HSS + nloc] = f2bf(v);
          }
        }
      }
    }
    __syncthreads();
    #pragma unroll
    for (int klc = 0; klc < 256; klc += 32) {
      FragB16 a2, b2;
      { const unsigned short* p = &s_hs[(mt2 * 16 + l15) * HSS + klc + lhi * 16];
        a2.q[0] = *(const uint4*)p;
        a2.q[1] = *(const uint4*)(p + 8); }
      { int n = nt2 * 16 + l15;
        const unsigned short* p = &We2t[n * Dd + g * 256 + klc + lhi * 16];
        b2.q[0] = *(const uint4*)p;
        b2.q[1] = *(const uint4*)(p + 8); }
      acc2 = __builtin_amdgcn_wmma_f32_16x16x32_bf16(
          false, a2.v, false, b2.v, (short)0, acc2, false, false);
    }
    __syncthreads();
  }

  // stage-2 epilogue: add gate-weighted be2, stage mix as bf16
  {
    int n = nt2 * 16 + l15;
    int mb = mt2 * 16 + lhi * 8;
    #pragma unroll
    for (int r = 0; r < 8; ++r) {
      float bmix = 0.f;
      #pragma unroll
      for (int e = 0; e < 8; ++e) bmix += s_gw[(mb + r) * 8 + e] * be2[e * Hh + n];
      s_mix[(mb + r) * MXS + n] = f2bf(acc2[r] + bmix);
    }
  }
  __syncthreads();

  // ---- stage 3: t = relu(mix @ Wc1 + bc1), M=64 N=64 K=128 ----
  if (w < 16) {
    v8f acc3 = {};
    int mt3 = w >> 2, nt3 = w & 3;
    #pragma unroll
    for (int kc = 0; kc < Hh; kc += 32) {
      FragB16 a3, b3;
      { const unsigned short* p = &s_mix[(mt3 * 16 + l15) * MXS + kc + lhi * 16];
        a3.q[0] = *(const uint4*)p;
        a3.q[1] = *(const uint4*)(p + 8); }
      { const unsigned short* p = &Wc1t[(nt3 * 16 + l15) * Hh + kc + lhi * 16];
        b3.q[0] = *(const uint4*)p;
        b3.q[1] = *(const uint4*)(p + 8); }
      acc3 = __builtin_amdgcn_wmma_f32_16x16x32_bf16(
          false, a3.v, false, b3.v, (short)0, acc3, false, false);
    }
    int n = nt3 * 16 + l15;
    int mb = mt3 * 16 + lhi * 8;
    #pragma unroll
    for (int r = 0; r < 8; ++r)
      s_t[(mb + r) * TS + n] = fmaxf(acc3[r] + bc1[n], 0.0f);
  }
  __syncthreads();

  // ---- final: out = t @ Wc2 + bc2, N=2 (VALU dot) ----
  if (tid < TM * 2) {
    int t = tid >> 1, c = tid & 1;
    float s = bc2[c];
    #pragma unroll 8
    for (int j = 0; j < 64; ++j) s += s_t[t * TS + j] * Wc2[j * 2 + c];
    out[(tok0 + t) * 2 + c] = s;
  }
}

// ---------------- launch ----------------
extern "C" void kernel_launch(void* const* d_in, const int* in_sizes, int n_in,
                              void* d_out, int out_size, void* d_ws, size_t ws_size,
                              hipStream_t stream) {
  const float* x   = (const float*)d_in[0];
  const float* We1 = (const float*)d_in[1];
  const float* be1 = (const float*)d_in[2];
  const float* We2 = (const float*)d_in[3];
  const float* be2 = (const float*)d_in[4];
  const float* Wg  = (const float*)d_in[5];
  const float* bg  = (const float*)d_in[6];
  const float* Wc1 = (const float*)d_in[7];
  const float* bc1 = (const float*)d_in[8];
  const float* Wc2 = (const float*)d_in[9];
  const float* bc2 = (const float*)d_in[10];
  float* out = (float*)d_out;

  unsigned short* We1t = (unsigned short*)d_ws;            // 2 MB
  unsigned short* We2t = We1t + 1024 * 1024;               // 256 KB
  unsigned short* Wc1t = We2t + 128 * 1024;                // 16 KB

  int prepN = 1024 * 1024 + 128 * 1024 + 64 * 128;
  prep_weights<<<(prepN + 255) / 256, 256, 0, stream>>>(We1, We2, Wc1, We1t, We2t, Wc1t);

  moe_fused<<<65536 / TM, 1024, 0, stream>>>(x, be1, be2, Wg, bg, bc1, Wc2, bc2,
                                             We1t, We2t, Wc1t, out);
}
